// DifferentPolicyNetwork_87737591923428
// MI455X (gfx1250) — matrix operand order repaired and checked by reading, not compiled, not run on hardware
//
#include <hip/hip_runtime.h>

// ---------------------------------------------------------------------------
// MoE policy head, fused, bf16 WMMA (gfx1250 / MI455X)
//   h1 = relu(state @ linear1[opt]);  h2 = relu(h1 @ linear2[opt])
//   mean = h2 @ mean_w[opt];  log_std = clip(h2 @ log_std_w[opt], -20, 2)
// ---------------------------------------------------------------------------

#define NOPT 8
#define NIN 128
#define SH 1024
#define HID 128
#define NACT 32
#define NBATCH 4096

typedef __bf16 v16bf __attribute__((ext_vector_type(16)));
typedef float  v8f   __attribute__((ext_vector_type(8)));

union FragU { unsigned int u[8]; v16bf v; };

__device__ __forceinline__ unsigned short f32_to_bf16_rne(float f) {
  unsigned int u = __float_as_uint(f);
  u += 0x7fffu + ((u >> 16) & 1u);   // round to nearest even
  return (unsigned short)(u >> 16);
}

// ---------------------------------------------------------------------------
// 1) zero per-expert counters
// ---------------------------------------------------------------------------
__global__ void zero_counts_kernel(int* counts) {
  if (threadIdx.x < NOPT) counts[threadIdx.x] = 0;
}

// ---------------------------------------------------------------------------
// 2) bucket batch rows by expert (order within bucket irrelevant: each row's
//    result is identical regardless of slot -> deterministic output)
// ---------------------------------------------------------------------------
__global__ void build_index_kernel(const int* __restrict__ option,
                                   int* __restrict__ counts,
                                   int* __restrict__ idx) {
  int b = blockIdx.x * blockDim.x + threadIdx.x;
  if (b >= NBATCH) return;
  int o = option[b] & 7;
  int pos = atomicAdd(&counts[o], 1);
  idx[o * NBATCH + pos] = b;
}

// ---------------------------------------------------------------------------
// 3) pack f32 weights W[o][K][N] -> bf16 WMMA B-fragments.
//    Fragment (o,ks,nt) = 32x16 bf16 block: K rows [ks*32, ks*32+32),
//    N cols [nt*16, nt*16+16). Per-lane slot j holds K = ks*32 + 16*(lane>>4)
//    + 2j (lo half) and +1 (hi half); N = nt*16 + (lane&15).
//    Memory: P[(((o*KS+ks)*NT+nt)*32 + lane)*8 + j] -> lane loads 2x b128.
// ---------------------------------------------------------------------------
__global__ void pack_weights_kernel(const float* __restrict__ W,
                                    unsigned int* __restrict__ P,
                                    int K, int N, int KSTEPS, int NTILES) {
  int gid = blockIdx.x * blockDim.x + threadIdx.x;
  int total = NOPT * KSTEPS * NTILES * 256;
  if (gid >= total) return;
  int j    = gid & 7;
  int lane = (gid >> 3) & 31;
  int f    = gid >> 8;                 // (o*KS + ks)*NT + nt
  int nt   = f % NTILES;
  int rest = f / NTILES;
  int ks   = rest % KSTEPS;
  int o    = rest / KSTEPS;
  int n = nt * 16 + (lane & 15);
  int k = ks * 32 + ((lane >> 4) << 4) + 2 * j;
  const float* w = W + ((size_t)o * K + k) * N + n;
  unsigned int lo = f32_to_bf16_rne(w[0]);
  unsigned int hi = f32_to_bf16_rne(w[N]);   // k+1
  P[gid] = lo | (hi << 16);
}

// ---------------------------------------------------------------------------
// Fragment helpers
// ---------------------------------------------------------------------------
__device__ __forceinline__ v16bf load_b_frag(const unsigned int* __restrict__ P,
                                             int fragIdx, int lane) {
  const uint4* p = (const uint4*)(P + (size_t)fragIdx * 256 + lane * 8);
  uint4 a = p[0];
  uint4 b = p[1];
  FragU f;
  f.u[0] = a.x; f.u[1] = a.y; f.u[2] = a.z; f.u[3] = a.w;
  f.u[4] = b.x; f.u[5] = b.y; f.u[6] = b.z; f.u[7] = b.w;
  return f.v;
}

// A fragment 16x32 bf16 from row-major bf16 LDS tile (rowStride in elements).
// lane<16: K = kBase + {0..15}; lane>=16: K = kBase + {16..31}.
__device__ __forceinline__ v16bf load_a_frag(const unsigned short* smem,
                                             int lane, int kBase, int rowStride) {
  int m    = lane & 15;
  int half = lane >> 4;
  const unsigned int* p = (const unsigned int*)(smem + (size_t)m * rowStride);
  int k0 = (kBase + half * 16) >> 1;       // word index, K pairs 0..3
  int k1 = (kBase + half * 16 + 8) >> 1;   // word index, K pairs 4..7
  FragU f;
  f.u[0] = p[k0 + 0]; f.u[1] = p[k0 + 1]; f.u[2] = p[k0 + 2]; f.u[3] = p[k0 + 3];
  f.u[4] = p[k1 + 0]; f.u[5] = p[k1 + 1]; f.u[6] = p[k1 + 2]; f.u[7] = p[k1 + 3];
  return f.v;
}

__device__ __forceinline__ v8f wmma_bf16(v16bf a, v16bf b, v8f c) {
  return __builtin_amdgcn_wmma_f32_16x16x32_bf16(
      /*neg_a=*/false, a, /*neg_b=*/false, b,
      /*c_mod=*/(short)0, c, /*reuse_a=*/false, /*reuse_b=*/false);
}

// ---------------------------------------------------------------------------
// 4) fused per-tile MoE kernel.  grid = (256 tiles, 8 experts), block = 256
//    (8 wave32). Each block: 16 gathered rows of one expert, fully fused.
// ---------------------------------------------------------------------------
__global__ __launch_bounds__(256, 1)
void moe_policy_kernel(const float* __restrict__ state,
                       const int* __restrict__ counts,
                       const int* __restrict__ idx,
                       const unsigned int* __restrict__ pL1,
                       const unsigned int* __restrict__ pL2,
                       const unsigned int* __restrict__ pMean,
                       const unsigned int* __restrict__ pLstd,
                       float* __restrict__ out) {
  const int e    = blockIdx.y;
  const int tile = blockIdx.x;
  const int cnt  = counts[e];
  if (tile * 16 >= cnt) return;          // block-uniform exit

  __shared__ __align__(16) unsigned short sState[16 * NIN];   // 4 KB bf16
  __shared__ __align__(16) unsigned short sH1[16 * SH];       // 32 KB bf16
  __shared__ __align__(16) unsigned short sH2[16 * HID];      // 4 KB bf16
  __shared__ int sRows[16];

  const int tid  = threadIdx.x;
  const int wave = tid >> 5;
  const int lane = tid & 31;
  const int half = lane >> 4;
  const int nlow = lane & 15;

  if (tid < 16) {
    int r = tile * 16 + tid;
    if (r > cnt - 1) r = cnt - 1;        // pad with last valid row (masked later)
    sRows[tid] = idx[e * NBATCH + r];
  }
  __syncthreads();

  // stage state tile [16][128] as bf16
  for (int i = tid; i < 16 * NIN; i += 256) {
    int m = i >> 7;
    int c = i & (NIN - 1);
    sState[i] = f32_to_bf16_rne(state[(size_t)sRows[m] * NIN + c]);
  }
  __syncthreads();

  // ---- GEMM1: [16,128] x [128,1024] -> h1, wave w owns N cols [w*128, w*128+128)
  {
    v16bf aF[4];
#pragma unroll
    for (int ks = 0; ks < 4; ++ks) aF[ks] = load_a_frag(sState, lane, ks * 32, NIN);
#pragma unroll
    for (int ntl = 0; ntl < 8; ++ntl) {
      int nt = wave * 8 + ntl;           // of 64 N-tiles
      v8f acc = {0.f, 0.f, 0.f, 0.f, 0.f, 0.f, 0.f, 0.f};
#pragma unroll
      for (int ks = 0; ks < 4; ++ks) {
        v16bf bF = load_b_frag(pL1, (e * 4 + ks) * 64 + nt, lane);
        acc = wmma_bf16(aF[ks], bF, acc);
      }
      int n = nt * 16 + nlow;
#pragma unroll
      for (int r = 0; r < 8; ++r) {
        float v = acc[r];
        v = v > 0.f ? v : 0.f;           // relu
        sH1[(size_t)(r + 8 * half) * SH + n] = f32_to_bf16_rne(v);
      }
    }
  }
  __syncthreads();

  // ---- GEMM2: [16,1024] x [1024,128] -> h2, wave w owns N-tile w (16 cols)
  {
    v8f acc = {0.f, 0.f, 0.f, 0.f, 0.f, 0.f, 0.f, 0.f};
#pragma unroll 4
    for (int ks = 0; ks < 32; ++ks) {
      v16bf aF = load_a_frag(sH1, lane, ks * 32, SH);
      v16bf bF = load_b_frag(pL2, (e * 32 + ks) * 8 + wave, lane);
      acc = wmma_bf16(aF, bF, acc);
    }
    int n = wave * 16 + nlow;
#pragma unroll
    for (int r = 0; r < 8; ++r) {
      float v = acc[r];
      v = v > 0.f ? v : 0.f;             // relu
      sH2[(size_t)(r + 8 * half) * HID + n] = f32_to_bf16_rne(v);
    }
  }
  __syncthreads();

  // ---- GEMM3/4: [16,128] x [128,32] heads. waves 0-1: mean, waves 2-3: log_std
  if (wave < 4) {                         // wave-uniform branch
    const int isLstd = wave >> 1;
    const int nt     = wave & 1;
    const unsigned int* P = isLstd ? pLstd : pMean;
    v16bf aF[4];
#pragma unroll
    for (int ks = 0; ks < 4; ++ks) aF[ks] = load_a_frag(sH2, lane, ks * 32, HID);
    v8f acc = {0.f, 0.f, 0.f, 0.f, 0.f, 0.f, 0.f, 0.f};
#pragma unroll
    for (int ks = 0; ks < 4; ++ks) {
      v16bf bF = load_b_frag(P, (e * 4 + ks) * 2 + nt, lane);
      acc = wmma_bf16(aF[ks], bF, acc);
    }
    int n = nt * 16 + nlow;
    size_t outOff = (size_t)isLstd * (NBATCH * NACT);
#pragma unroll
    for (int r = 0; r < 8; ++r) {
      int m = r + 8 * half;
      if (tile * 16 + m < cnt) {          // mask padded rows
        float v = acc[r];
        if (isLstd) v = fminf(fmaxf(v, -20.f), 2.f);
        out[outOff + (size_t)sRows[m] * NACT + n] = v;
      }
    }
  }
}

// ---------------------------------------------------------------------------
// launch
// ---------------------------------------------------------------------------
extern "C" void kernel_launch(void* const* d_in, const int* in_sizes, int n_in,
                              void* d_out, int out_size, void* d_ws, size_t ws_size,
                              hipStream_t stream) {
  const float* state   = (const float*)d_in[0];
  const int*   option  = (const int*)d_in[1];
  const float* linear1 = (const float*)d_in[2];
  const float* linear2 = (const float*)d_in[3];
  const float* mean_w  = (const float*)d_in[4];
  const float* lstd_w  = (const float*)d_in[5];
  float* out = (float*)d_out;

  char* ws = (char*)d_ws;
  // workspace layout (bytes)
  int*          counts = (int*)(ws + 0);                       //     32 B
  int*          idx    = (int*)(ws + 1024);                    //  128 KB
  unsigned int* pL1    = (unsigned int*)(ws + 0x00030000);     //   2 MB  (8*4*64*256 u32)
  unsigned int* pL2    = (unsigned int*)(ws + 0x00230000);     //   2 MB  (8*32*8*256 u32)
  unsigned int* pMean  = (unsigned int*)(ws + 0x00430000);     //  64 KB  (8*4*2*256 u32)
  unsigned int* pLstd  = (unsigned int*)(ws + 0x00440000);     //  64 KB

  zero_counts_kernel<<<1, 32, 0, stream>>>(counts);
  build_index_kernel<<<NBATCH / 256, 256, 0, stream>>>(option, counts, idx);

  // pack weights: (K, N, KSTEPS, NTILES)
  pack_weights_kernel<<<(NOPT * 4 * 64 * 256) / 256, 256, 0, stream>>>(
      linear1, pL1, NIN, SH, 4, 64);
  pack_weights_kernel<<<(NOPT * 32 * 8 * 256) / 256, 256, 0, stream>>>(
      linear2, pL2, SH, HID, 32, 8);
  pack_weights_kernel<<<(NOPT * 4 * 2 * 256) / 256, 256, 0, stream>>>(
      mean_w, pMean, HID, NACT, 4, 2);
  pack_weights_kernel<<<(NOPT * 4 * 2 * 256) / 256, 256, 0, stream>>>(
      lstd_w, pLstd, HID, NACT, 4, 2);

  dim3 grid(NBATCH / 16, NOPT);   // blocks past a bucket's tile count exit early
  moe_policy_kernel<<<grid, 256, 0, stream>>>(
      state, counts, idx, pL1, pL2, pMean, pLstd, out);
}
